// ChannelAtten_20486994001925
// MI455X (gfx1250) — matrix operand order repaired and checked by reading, not compiled, not run on hardware
//
#include <hip/hip_runtime.h>

// ---------- types ----------
typedef __bf16 v16bf __attribute__((ext_vector_type(16)));
typedef __bf16 v8bf  __attribute__((ext_vector_type(8)));
typedef float  v8f   __attribute__((ext_vector_type(8)));

#define DIM      192
#define QKV_DIM  576
#define HEAD_DIM 48
#define NH       4
#define BATCH    4
#define HW       65536   // 256*256
#define SIDE     256

static __device__ __forceinline__ v16bf cat8(v8bf lo, v8bf hi) {
  return __builtin_shufflevector(lo, hi, 0,1,2,3,4,5,6,7,8,9,10,11,12,13,14,15);
}

// ============================================================
// k0: convert qkv weight [576*192] f32 -> bf16
// ============================================================
__global__ void k0_prep(const float* __restrict__ w, __bf16* __restrict__ wbf, int n) {
  int i = blockIdx.x * 256 + threadIdx.x;
  if (i < n) wbf[i] = (__bf16)w[i];
}

// ============================================================
// k1: qkv = Wqkv(576x192) @ x(192xHW) + b  -> bf16 [B][576][HW]
// grid (HW/128, B), block 256 (8 waves). WMMA bf16.
// A-frag layout (16-bit A 16x32): lanes0-15 row=M0+l, K {0..7,16..23};
// lanes16-31 row=M0+l-16, K {8..15,24..31}.
// B-frag (16-bit B 32x16): lanes0-15 col=N0+l, K {0..7,16..23}; lanes16-31 K {8..15,24..31}.
// C (f32 16x16): vgpr r -> row M0 + r + 8*(lane>=16), col N0 + lane%16.
// ============================================================
__global__ __launch_bounds__(256) void k1_qkv(
    const float* __restrict__ x, const __bf16* __restrict__ Wbf,
    const float* __restrict__ b_qkv, __bf16* __restrict__ qkv) {
  const int b  = blockIdx.y;
  const int s0 = blockIdx.x * 128;
  __shared__ __align__(16) __bf16 Xs[128][200];   // [spatial n][channel k], padded

  { // stage x tile -> LDS (transpose, f32->bf16)
    int n  = threadIdx.x & 127;
    int c0 = threadIdx.x >> 7;
    const float* xb = x + (long)b * DIM * HW + s0 + n;
    for (int c = c0; c < DIM; c += 2)
      Xs[n][c] = (__bf16)xb[(long)c * HW];
  }
  __syncthreads();

  const int lane = threadIdx.x & 31;
  const int wave = threadIdx.x >> 5;
  const int half = lane >> 4;
  const int lr   = lane & 15;

  for (int mt = wave; mt < QKV_DIM / 16; mt += 8) {
    const int m0 = mt * 16;
    // 6 A fragments (K = 192 = 6*32), register resident
    v16bf a[6];
    const __bf16* wrow = Wbf + (m0 + lr) * DIM + half * 8;
#pragma unroll
    for (int kk = 0; kk < 6; kk++) {
      v8bf lo = *(const v8bf*)(wrow + kk * 32);
      v8bf hi = *(const v8bf*)(wrow + kk * 32 + 16);
      a[kk] = cat8(lo, hi);
    }
    float bias[8];
#pragma unroll
    for (int r = 0; r < 8; r++) bias[r] = b_qkv[m0 + half * 8 + r];

    for (int nt = 0; nt < 8; nt++) {
      const int n0 = nt * 16;
      v8f c = {};
#pragma unroll
      for (int kk = 0; kk < 6; kk++) {
        const int kb = kk * 32 + half * 8;
        v8bf lo = *(const v8bf*)&Xs[n0 + lr][kb];
        v8bf hi = *(const v8bf*)&Xs[n0 + lr][kb + 16];
        v16bf bf = cat8(lo, hi);
        c = __builtin_amdgcn_wmma_f32_16x16x32_bf16(false, a[kk], false, bf,
                                                    (short)0, c, false, false);
      }
#pragma unroll
      for (int r = 0; r < 8; r++) {
        const int row = m0 + half * 8 + r;
        qkv[((long)b * QKV_DIM + row) * HW + s0 + n0 + lr] = (__bf16)(c[r] + bias[r]);
      }
    }
  }
}

// ============================================================
// k2: depthwise 3x3 (pad 1) on qkv bf16 -> dwqkv bf16.
// Also accumulates sum(y^2) per (b, channel) for q (ch<192) and k (192..383).
// grid (64 row-tiles, 576, 4), block 256 (4 rows x 256 cols per block).
// ============================================================
__global__ __launch_bounds__(256) void k2_dwconv(
    const __bf16* __restrict__ qkv, const float* __restrict__ w_dw,
    const float* __restrict__ b_dw, __bf16* __restrict__ dwqkv,
    float* __restrict__ norm2) {
  const int b = blockIdx.z, ch = blockIdx.y, tile = blockIdx.x;
  const __bf16* in  = qkv   + ((long)b * QKV_DIM + ch) * HW;
  __bf16*       out = dwqkv + ((long)b * QKV_DIM + ch) * HW;
  float w[9];
#pragma unroll
  for (int i = 0; i < 9; i++) w[i] = w_dw[ch * 9 + i];
  const float bias = b_dw[ch];
  const int y0 = tile * 4;

  float acc2 = 0.f;
  for (int p = threadIdx.x; p < 4 * SIDE; p += 256) {
    const int ly = p >> 8, xx = p & 255;
    const int y = y0 + ly;
    float s = bias;
#pragma unroll
    for (int dy = -1; dy <= 1; dy++) {
      const int yy = y + dy;
      if (yy < 0 || yy > SIDE - 1) continue;
#pragma unroll
      for (int dx = -1; dx <= 1; dx++) {
        const int xc = xx + dx;
        if (xc < 0 || xc > SIDE - 1) continue;
        s += w[(dy + 1) * 3 + (dx + 1)] * (float)in[yy * SIDE + xc];
      }
    }
    out[y * SIDE + xx] = (__bf16)s;
    acc2 += s * s;
  }
  __shared__ float red[256];
  red[threadIdx.x] = acc2;
  __syncthreads();
  for (int off = 128; off > 0; off >>= 1) {
    if (threadIdx.x < (unsigned)off) red[threadIdx.x] += red[threadIdx.x + off];
    __syncthreads();
  }
  if (threadIdx.x == 0 && ch < 2 * DIM) atomicAdd(&norm2[b * 2 * DIM + ch], red[0]);
}

// ============================================================
// k3: per (b, head): G[48][48] += Q_chunk @ K_chunk^T   (WMMA bf16, f32 acc)
// grid (16 S-chunks of 4096, NH, B), block 288 = 9 waves, one 16x16 tile each.
// ============================================================
__global__ __launch_bounds__(288) void k3_gram(
    const __bf16* __restrict__ dwqkv, float* __restrict__ G) {
  const int b = blockIdx.z, h = blockIdx.y, chunk = blockIdx.x;
  const int wave = threadIdx.x >> 5, lane = threadIdx.x & 31;
  const int mt = wave / 3, nt = wave % 3;
  const int half = lane >> 4, lr = lane & 15;

  const __bf16* Q = dwqkv + ((long)b * QKV_DIM + h * HEAD_DIM) * HW;
  const __bf16* K = dwqkv + ((long)b * QKV_DIM + DIM + h * HEAD_DIM) * HW;
  const long sbase = (long)chunk * 4096;
  const __bf16* qrow = Q + (long)(mt * 16 + lr) * HW + sbase + half * 8;
  const __bf16* krow = K + (long)(nt * 16 + lr) * HW + sbase + half * 8;

  v8f c = {};
  for (int s = 0; s < 4096; s += 32) {
    __builtin_prefetch(qrow + s + 512, 0, 1);
    __builtin_prefetch(krow + s + 512, 0, 1);
    v8bf alo = *(const v8bf*)(qrow + s);
    v8bf ahi = *(const v8bf*)(qrow + s + 16);
    v8bf blo = *(const v8bf*)(krow + s);
    v8bf bhi = *(const v8bf*)(krow + s + 16);
    c = __builtin_amdgcn_wmma_f32_16x16x32_bf16(false, cat8(alo, ahi), false,
                                                cat8(blo, bhi), (short)0, c,
                                                false, false);
  }
#pragma unroll
  for (int r = 0; r < 8; r++) {
    const int d = mt * 16 + half * 8 + r;
    const int e = nt * 16 + lr;
    atomicAdd(&G[(((b * NH) + h) * HEAD_DIM + d) * HEAD_DIM + e], c[r]);
  }
}

// ============================================================
// k4: attn = softmax(G / (|q||k|) * scale); M = Wproj @ blockdiag(attn) -> bf16
// grid (B), block 192.
// ============================================================
__global__ __launch_bounds__(192) void k4_softmax_M(
    const float* __restrict__ G, const float* __restrict__ norm2,
    const float* __restrict__ w_proj, __bf16* __restrict__ Mbf) {
  const int b = blockIdx.x;
  const int t = threadIdx.x;                 // 0..191
  __shared__ float A[DIM][HEAD_DIM];         // A[h*48+d][e]
  const float scale = rsqrtf((float)HEAD_DIM);
  {
    const int h = t / HEAD_DIM, d = t % HEAD_DIM;
    const float nq = fmaxf(sqrtf(norm2[b * 2 * DIM + h * HEAD_DIM + d]), 1e-12f);
    float row[HEAD_DIM];
    float mx = -3.4e38f;
#pragma unroll 4
    for (int e = 0; e < HEAD_DIM; e++) {
      const float nk = fmaxf(sqrtf(norm2[b * 2 * DIM + DIM + h * HEAD_DIM + e]), 1e-12f);
      const float v = G[(((b * NH) + h) * HEAD_DIM + d) * HEAD_DIM + e] / (nq * nk) * scale;
      row[e] = v;
      mx = fmaxf(mx, v);
    }
    float s = 0.f;
#pragma unroll 4
    for (int e = 0; e < HEAD_DIM; e++) { row[e] = __expf(row[e] - mx); s += row[e]; }
    const float inv = 1.f / s;
#pragma unroll 4
    for (int e = 0; e < HEAD_DIM; e++) A[t][e] = row[e] * inv;
  }
  __syncthreads();
  // M[t][h*48+e] = sum_d Wp[t][h*48+d] * A[h*48+d][e]
  for (int cp = 0; cp < DIM; cp++) {
    const int h = cp / HEAD_DIM, e = cp % HEAD_DIM;
    float s = 0.f;
#pragma unroll 8
    for (int d = 0; d < HEAD_DIM; d++)
      s += w_proj[t * DIM + h * HEAD_DIM + d] * A[h * HEAD_DIM + d][e];
    Mbf[((long)b * DIM + t) * DIM + cp] = (__bf16)s;
  }
}

// ============================================================
// k5: out = M_b(192x192) @ V_b(192xHW) + b_proj  -> f32 d_out
// grid (HW/128, B), block 256. V = dwqkv channels 384..575.
// ============================================================
__global__ __launch_bounds__(256) void k5_out(
    const __bf16* __restrict__ dwqkv, const __bf16* __restrict__ Mbf,
    const float* __restrict__ b_proj, float* __restrict__ out) {
  const int b  = blockIdx.y;
  const int s0 = blockIdx.x * 128;
  __shared__ __align__(16) __bf16 Vs[128][200];

  { // stage V tile (already bf16) -> LDS transposed
    int n  = threadIdx.x & 127;
    int c0 = threadIdx.x >> 7;
    const __bf16* vb = dwqkv + ((long)b * QKV_DIM + 2 * DIM) * HW + s0 + n;
    for (int c = c0; c < DIM; c += 2)
      Vs[n][c] = vb[(long)c * HW];
  }
  __syncthreads();

  const int lane = threadIdx.x & 31;
  const int wave = threadIdx.x >> 5;
  const int half = lane >> 4;
  const int lr   = lane & 15;

  for (int mt = wave; mt < DIM / 16; mt += 8) {
    const int m0 = mt * 16;
    v16bf a[6];
    const __bf16* wrow = Mbf + ((long)b * DIM + m0 + lr) * DIM + half * 8;
#pragma unroll
    for (int kk = 0; kk < 6; kk++) {
      v8bf lo = *(const v8bf*)(wrow + kk * 32);
      v8bf hi = *(const v8bf*)(wrow + kk * 32 + 16);
      a[kk] = cat8(lo, hi);
    }
    float bias[8];
#pragma unroll
    for (int r = 0; r < 8; r++) bias[r] = b_proj[m0 + half * 8 + r];

    for (int nt = 0; nt < 8; nt++) {
      const int n0 = nt * 16;
      v8f c = {};
#pragma unroll
      for (int kk = 0; kk < 6; kk++) {
        const int kb = kk * 32 + half * 8;
        v8bf lo = *(const v8bf*)&Vs[n0 + lr][kb];
        v8bf hi = *(const v8bf*)&Vs[n0 + lr][kb + 16];
        c = __builtin_amdgcn_wmma_f32_16x16x32_bf16(false, a[kk], false,
                                                    cat8(lo, hi), (short)0, c,
                                                    false, false);
      }
#pragma unroll
      for (int r = 0; r < 8; r++) {
        const int row = m0 + half * 8 + r;
        out[((long)b * DIM + row) * HW + s0 + n0 + lr] = c[r] + bias[r];
      }
    }
  }
}

// ============================================================
// launch
// ============================================================
static inline size_t alignup(size_t v) { return (v + 255) & ~(size_t)255; }

extern "C" void kernel_launch(void* const* d_in, const int* in_sizes, int n_in,
                              void* d_out, int out_size, void* d_ws, size_t ws_size,
                              hipStream_t stream) {
  const float* x      = (const float*)d_in[0];
  const float* w_qkv  = (const float*)d_in[1];
  const float* b_qkv  = (const float*)d_in[2];
  const float* w_dw   = (const float*)d_in[3];
  const float* b_dw   = (const float*)d_in[4];
  const float* w_proj = (const float*)d_in[5];
  const float* b_proj = (const float*)d_in[6];
  float* out = (float*)d_out;

  char* p = (char*)d_ws;
  __bf16* Wbf    = (__bf16*)p; p += alignup((size_t)QKV_DIM * DIM * 2);
  __bf16* qkvbf  = (__bf16*)p; p += alignup((size_t)BATCH * QKV_DIM * HW * 2);
  __bf16* dwqkv  = (__bf16*)p; p += alignup((size_t)BATCH * QKV_DIM * HW * 2);
  float*  norm2  = (float*)p;  p += (size_t)BATCH * 2 * DIM * 4;          // keep
  float*  G      = (float*)p;  p += (size_t)BATCH * NH * HEAD_DIM * HEAD_DIM * 4; // contiguous
  __bf16* Mbf    = (__bf16*)p; p += alignup((size_t)BATCH * DIM * DIM * 2);

  // zero the atomic accumulators (norm2 + G contiguous)
  hipMemsetAsync(norm2, 0,
                 (size_t)(BATCH * 2 * DIM + BATCH * NH * HEAD_DIM * HEAD_DIM) * 4,
                 stream);

  k0_prep<<<(QKV_DIM * DIM + 255) / 256, 256, 0, stream>>>(w_qkv, Wbf, QKV_DIM * DIM);
  k1_qkv<<<dim3(HW / 128, BATCH), 256, 0, stream>>>(x, Wbf, b_qkv, qkvbf);
  k2_dwconv<<<dim3(SIDE / 4, QKV_DIM, BATCH), 256, 0, stream>>>(qkvbf, w_dw, b_dw,
                                                                dwqkv, norm2);
  k3_gram<<<dim3(16, NH, BATCH), 288, 0, stream>>>(dwqkv, G);
  k4_softmax_M<<<BATCH, DIM, 0, stream>>>(G, norm2, w_proj, Mbf);
  k5_out<<<dim3(HW / 128, BATCH), 256, 0, stream>>>(dwqkv, Mbf, b_proj, out);
}
